// LSTM_29910152249709
// MI455X (gfx1250) — matrix operand order repaired and checked by reading, not compiled, not run on hardware
//
#include <hip/hip_runtime.h>

// ---------------------------------------------------------------------------
// MI455X (gfx1250) LSTM.
//   Phase 1 (prep):    fp32 -> bf16 weight/X conversion, bias stacking, init.
//   Phase 2 (xgemm):   Xg[t,n] = Wx @ x_t + bg  via v_wmma_f32_16x16x32_bf16.
//   Phase 3 (lstm_rec): persistent kernel, 64 WGs, 2 grid barriers per step,
//                       WMMA GEMV (A = replicated h/c rows, B = row-major W).
//   All WMMA stages use two independent accumulator chains to break the
//   WMMA->WMMA C-operand dependency chain (ISA 7.12.1 hazards).
// ---------------------------------------------------------------------------

typedef __attribute__((ext_vector_type(16))) __bf16 v16bf;
typedef __attribute__((ext_vector_type(8)))  __bf16 v8bf;
typedef __attribute__((ext_vector_type(8)))  float  v8f;

#define T_STEPS   8192
#define IN_DIM    300
#define KPAD      320          // IN_DIM padded to multiple of 32
#define CELL      1024
#define GDIM      4096         // 4*CELL stacked gate rows
#define DIN       1324         // IN_DIM + CELL
#define NWG       64           // persistent workgroups (16 cells each)

__device__ __forceinline__ float sigf(float x) { return 1.f / (1.f + __expf(-x)); }

// Pack A-operand for 16x16x32 bf16 WMMA with all 16 rows replicated from a
// vector chunk v[k0..k0+31]. Lane l needs v[koff+0..15]
// with koff = (l>>4)*16.  p0 = &v[k0+koff].
__device__ __forceinline__ v16bf packA(const __bf16* p0) {
    v8bf a0 = *(const v8bf*)(p0);
    v8bf a1 = *(const v8bf*)(p0 + 8);
    v16bf A;
#pragma unroll
    for (int i = 0; i < 8; ++i) { A[i] = a0[i]; A[i + 8] = a1[i]; }
    return A;
}

__device__ __forceinline__ v8f wmma_bf16(v16bf A, v16bf B, v8f C) {
    return __builtin_amdgcn_wmma_f32_16x16x32_bf16(false, A, false, B,
                                                   (short)0, C, false, false);
}

// -------------------------- Phase 1: prep ----------------------------------
__global__ void lstm_prep(const float* __restrict__ X,  const float* __restrict__ h0,
                          const float* __restrict__ Wf, const float* __restrict__ Wts,
                          const float* __restrict__ Wtt,const float* __restrict__ Wos,
                          const float* __restrict__ bf_,const float* __restrict__ bts,
                          const float* __restrict__ btt,const float* __restrict__ bos,
                          const float* __restrict__ Wot,
                          __bf16* Xbf, __bf16* Wgxbf, __bf16* Whbf, __bf16* Wotbf,
                          float* bg, __bf16* hbuf, int* bar)
{
    const long NXBF = (long)T_STEPS * KPAD;
    const long NWGX = (long)GDIM * KPAD;
    const long NWH  = (long)GDIM * CELL;
    const long NWOT = (long)CELL * CELL;
    const long TOTAL = NXBF + NWGX + NWH + NWOT + GDIM + CELL + 16;
    long stride = (long)gridDim.x * blockDim.x;
    for (long idx = (long)blockIdx.x * blockDim.x + threadIdx.x; idx < TOTAL; idx += stride) {
        long i = idx;
        if (i < NXBF) {                                  // X -> bf16, pad K
            long t = i / KPAD; int k = (int)(i % KPAD);
            Xbf[i] = (__bf16)(k < IN_DIM ? X[t * IN_DIM + k] : 0.f);
            continue;
        }
        i -= NXBF;
        if (i < NWGX) {                                  // Wg[:, :300] -> bf16, pad K
            int r = (int)(i / KPAD), k = (int)(i % KPAD);
            const float* W = (r < 1024) ? Wf : (r < 2048) ? Wts : (r < 3072) ? Wtt : Wos;
            int rr = r & 1023;
            Wgxbf[i] = (__bf16)(k < IN_DIM ? W[(long)rr * DIN + k] : 0.f);
            continue;
        }
        i -= NWGX;
        if (i < NWH) {                                   // Wg[:, 300:1324] -> bf16
            int r = (int)(i / CELL), k = (int)(i % CELL);
            const float* W = (r < 1024) ? Wf : (r < 2048) ? Wts : (r < 3072) ? Wtt : Wos;
            int rr = r & 1023;
            Whbf[i] = (__bf16)W[(long)rr * DIN + IN_DIM + k];
            continue;
        }
        i -= NWH;
        if (i < NWOT) { Wotbf[i] = (__bf16)Wot[i]; continue; }
        i -= NWOT;
        if (i < GDIM) {                                  // stacked gate bias
            const float* b = (i < 1024) ? bf_ : (i < 2048) ? bts : (i < 3072) ? btt : bos;
            bg[i] = b[i & 1023];
            continue;
        }
        i -= GDIM;
        if (i < CELL) { hbuf[i] = (__bf16)h0[i]; continue; }
        i -= CELL;
        bar[i] = 0;                                      // barrier state
    }
}

// ------------------ Phase 2: Xg = X @ Wx^T + bg (WMMA GEMM) ----------------
// One 16x16 output tile per wave; K = 320 -> 10 WMMAs in 2 chains of 5.
__global__ void __launch_bounds__(256) lstm_xgemm(const __bf16* __restrict__ Xbf,
                                                  const __bf16* __restrict__ Wgxbf,
                                                  const float* __restrict__ bg,
                                                  __bf16* __restrict__ Xg)
{
    int wave = threadIdx.x >> 5, lane = threadIdx.x & 31;
    int tile = blockIdx.x * 8 + wave;          // 512*256 = 131072 tiles
    int nt = tile & 255, mt = tile >> 8;
    int t0 = mt * 16, n0 = nt * 16;
    int lr = lane & 15, lh = lane >> 4;

    const __bf16* arow = Xbf   + (size_t)(t0 + lr) * KPAD + lh * 16;
    const __bf16* brow = Wgxbf + (size_t)(n0 + lr) * KPAD + lh * 16;

    v8f acc0 = {}, acc1 = {};
#pragma unroll
    for (int kt = 0; kt < 5; ++kt) {
        acc0 = wmma_bf16(packA(arow + (2 * kt) * 32),
                         *(const v16bf*)(brow + (2 * kt) * 32), acc0);
        acc1 = wmma_bf16(packA(arow + (2 * kt + 1) * 32),
                         *(const v16bf*)(brow + (2 * kt + 1) * 32), acc1);
    }
    float bias = bg[n0 + lr];
    int mbase = t0 + lh * 8;
#pragma unroll
    for (int r = 0; r < 8; ++r)
        Xg[(size_t)(mbase + r) * GDIM + (n0 + lr)] = (__bf16)(acc0[r] + acc1[r] + bias);
}

// ------------------------ device-scope grid barrier ------------------------
__device__ __forceinline__ void grid_barrier(int* cnt, int* gen)
{
    __builtin_amdgcn_fence(__ATOMIC_RELEASE, "agent");   // publish our writes
    __syncthreads();
    if (threadIdx.x == 0) {
        int g = __hip_atomic_load(gen, __ATOMIC_RELAXED, __HIP_MEMORY_SCOPE_AGENT);
        int prev = __hip_atomic_fetch_add(cnt, 1, __ATOMIC_ACQ_REL, __HIP_MEMORY_SCOPE_AGENT);
        if (prev == NWG - 1) {
            __hip_atomic_store(cnt, 0, __ATOMIC_RELAXED, __HIP_MEMORY_SCOPE_AGENT);
            __hip_atomic_store(gen, g + 1, __ATOMIC_RELEASE, __HIP_MEMORY_SCOPE_AGENT);
        } else {
            while (__hip_atomic_load(gen, __ATOMIC_ACQUIRE, __HIP_MEMORY_SCOPE_AGENT) == g)
                __builtin_amdgcn_s_sleep(2);
        }
    }
    __syncthreads();
    __builtin_amdgcn_fence(__ATOMIC_ACQUIRE, "agent");   // see others' writes
}

// -------------------- Phase 3: persistent recurrence -----------------------
// WG w owns cells [w*16, w*16+16). 8 waves:
//   stage 1: wave = kh*4+g computes gate g over K-half kh (16 WMMAs, 2 chains).
//   stage 2: wave w covers Wot K-range [w*128, w*128+128) (4 WMMAs, 2 chains).
// Weights stream from L2 (10.5 MB/step across 64 WGs << 192 MB L2).
__global__ void __launch_bounds__(256) lstm_rec(const __bf16* __restrict__ Xg,
                                                const __bf16* __restrict__ Whbf,
                                                const __bf16* __restrict__ Wotbf,
                                                const float* __restrict__ c0,
                                                const float* __restrict__ bot,
                                                __bf16* hbuf, __bf16* cbuf,
                                                int* bar, float* __restrict__ out)
{
    __shared__ float part[8][16];
    __shared__ float part2[8][16];
    __shared__ float c_sh[16];
    __shared__ float os_sh[16];

    const int wg = blockIdx.x, tid = threadIdx.x;
    const int wave = tid >> 5, lane = tid & 31;
    const int lr = lane & 15, lh = lane >> 4;
    const int g = wave & 3, kh = wave >> 2;

    if (tid < 16) c_sh[tid] = c0[wg * 16 + tid];
    __syncthreads();

    // Row-major B-tile bases (each lane: 16 contiguous bf16 of row n = lr).
    const __bf16* B1 = Whbf  + (size_t)(g * 1024 + wg * 16 + lr) * CELL + kh * 512 + lh * 16;
    const __bf16* B2 = Wotbf + (size_t)(wg * 16 + lr) * CELL + wave * 128 + lh * 16;
    const __bf16* hp_base = hbuf + kh * 512 + lh * 16;
    const __bf16* cp_base = cbuf + wave * 128 + lh * 16;

    for (int t = 0; t < T_STEPS; ++t) {
        // ---- stage 1: gate GEMV slice (Wh @ h), K-half per wave ----
        v8f acc0 = {}, acc1 = {};
#pragma unroll
        for (int kk = 0; kk < 8; ++kk) {
            acc0 = wmma_bf16(packA(hp_base + (2 * kk) * 32),
                             *(const v16bf*)(B1 + (2 * kk) * 32), acc0);
            acc1 = wmma_bf16(packA(hp_base + (2 * kk + 1) * 32),
                             *(const v16bf*)(B1 + (2 * kk + 1) * 32), acc1);
        }
        if (lane < 16) part[wave][lane] = acc0[0] + acc1[0];   // row-replicated D
        __syncthreads();

        if (tid < 16) {                                    // gate math, 16 cells
            int n = tid, row = wg * 16 + n;
            const __bf16* xg = Xg + (size_t)t * GDIM + row;
            float f_  = part[0][n] + part[4][n] + (float)xg[0];
            float ts_ = part[1][n] + part[5][n] + (float)xg[1024];
            float tt_ = part[2][n] + part[6][n] + (float)xg[2048];
            float os_ = part[3][n] + part[7][n] + (float)xg[3072];
            float cn = sigf(f_) * c_sh[n] + sigf(ts_) * tanhf(tt_);
            c_sh[n] = cn;
            os_sh[n] = os_;
            cbuf[row] = (__bf16)cn;                        // broadcast c_new
        }
        grid_barrier(&bar[0], &bar[1]);

        // ---- stage 2: Wot @ c_new, K-range per wave ----
        v8f acc2 = {}, acc3 = {};
#pragma unroll
        for (int kk = 0; kk < 2; ++kk) {
            acc2 = wmma_bf16(packA(cp_base + (2 * kk) * 32),
                             *(const v16bf*)(B2 + (2 * kk) * 32), acc2);
            acc3 = wmma_bf16(packA(cp_base + (2 * kk + 1) * 32),
                             *(const v16bf*)(B2 + (2 * kk + 1) * 32), acc3);
        }
        if (lane < 16) part2[wave][lane] = acc2[0] + acc3[0];
        __syncthreads();

        if (tid < 16) {
            int n = tid, row = wg * 16 + n;
            float o = bot[row];
#pragma unroll
            for (int w = 0; w < 8; ++w) o += part2[w][n];
            float hn = sigf(os_sh[n]) * tanhf(o);
            hbuf[row] = (__bf16)hn;                        // broadcast h_new
            out[(size_t)t * CELL + row] = hn;              // hs[t]
        }
        grid_barrier(&bar[0], &bar[1]);
    }
}

// ---------------------------------------------------------------------------
extern "C" void kernel_launch(void* const* d_in, const int* in_sizes, int n_in,
                              void* d_out, int out_size, void* d_ws, size_t ws_size,
                              hipStream_t stream)
{
    const float* X   = (const float*)d_in[0];
    const float* h0  = (const float*)d_in[1];
    const float* c0  = (const float*)d_in[2];
    const float* Wf  = (const float*)d_in[3];
    const float* bf_ = (const float*)d_in[4];
    const float* Wts = (const float*)d_in[5];
    const float* bts = (const float*)d_in[6];
    const float* Wtt = (const float*)d_in[7];
    const float* btt = (const float*)d_in[8];
    const float* Wos = (const float*)d_in[9];
    const float* bos = (const float*)d_in[10];
    const float* Wot = (const float*)d_in[11];
    const float* bot = (const float*)d_in[12];

    char* p = (char*)d_ws;                                  // all chunks 32B-multiple
    __bf16* Xg    = (__bf16*)p; p += (size_t)T_STEPS * GDIM * 2;   // 64 MB
    __bf16* Whbf  = (__bf16*)p; p += (size_t)GDIM * CELL * 2;      // 8 MB
    __bf16* Wotbf = (__bf16*)p; p += (size_t)CELL * CELL * 2;      // 2 MB
    __bf16* Wgxbf = (__bf16*)p; p += (size_t)GDIM * KPAD * 2;      // 2.5 MB
    __bf16* Xbf   = (__bf16*)p; p += (size_t)T_STEPS * KPAD * 2;   // 5 MB
    float*  bg    = (float*) p; p += (size_t)GDIM * 4;
    __bf16* hbuf  = (__bf16*)p; p += (size_t)CELL * 2;
    __bf16* cbuf  = (__bf16*)p; p += (size_t)CELL * 2;
    int*    bar   = (int*)   p; p += 64;

    lstm_prep<<<8192, 256, 0, stream>>>(X, h0, Wf, Wts, Wtt, Wos,
                                        bf_, bts, btt, bos, Wot,
                                        Xbf, Wgxbf, Whbf, Wotbf, bg, hbuf, bar);

    lstm_xgemm<<<16384, 256, 0, stream>>>(Xbf, Wgxbf, bg, Xg);

    lstm_rec<<<NWG, 256, 0, stream>>>(Xg, Whbf, Wotbf, c0, bot,
                                      hbuf, cbuf, bar, (float*)d_out);
}
